// CVFusionProposalRefiner_62878321214200
// MI455X (gfx1250) — matrix operand order repaired and checked by reading, not compiled, not run on hardware
//
#include <hip/hip_runtime.h>
#include <math.h>

// ---------------------------------------------------------------------------
// CDNA5 / gfx1250 implementation of the CVFusionProposalRefiner reference.
// Shapes (compile-time): B=4, Q=16384, S=4, QD=FD=HD=256, H=W=128.
// GEMMs: v_wmma_f32_16x16x32_bf16, 128x128 block tiles, 8 wave32/block,
// double-buffered LDS, async global->LDS staging (ASYNCcnt) for bf16 tiles.
// ---------------------------------------------------------------------------

typedef __attribute__((ext_vector_type(16))) __bf16 v16bf;
typedef __attribute__((ext_vector_type(8)))  float  v8f;

union BfFrag { v16bf v; unsigned short u[16]; };

__device__ __forceinline__ unsigned short f32_to_bf16(float f) {
  union { float f; unsigned int u; } v; v.f = f;
  unsigned int u = v.u;
  u += 0x7FFFu + ((u >> 16) & 1u);      // round-to-nearest-even
  return (unsigned short)(u >> 16);
}
__device__ __forceinline__ float bf16_to_f32(unsigned short h) {
  union { unsigned int u; float f; } v; v.u = ((unsigned int)h) << 16;
  return v.f;
}
__device__ __forceinline__ unsigned int pack_bf16x2(float lo, float hi) {
  return (unsigned int)f32_to_bf16(lo) | ((unsigned int)f32_to_bf16(hi) << 16);
}

// async b128 copy: global -> LDS, tracked by ASYNCcnt (CDNA5 §10 async ops)
__device__ __forceinline__ void async_copy_b128(void* lds_ptr, const void* gptr) {
  const unsigned int       lds  = (unsigned int)(size_t)lds_ptr;  // LDS byte offset
  const unsigned long long addr = (unsigned long long)(size_t)gptr;
  asm volatile("global_load_async_to_lds_b128 %0, %1, off"
               :: "v"(lds), "v"(addr) : "memory");
}
__device__ __forceinline__ void async_wait_all() {
  asm volatile("s_wait_asynccnt 0x0" ::: "memory");
}

// ---------------------------------------------------------------------------
// Weight transpose + f32 -> bf16 : dst[n*K + k] = bf16(src[k*N + n])
// ---------------------------------------------------------------------------
__global__ __launch_bounds__(256) void transpose_bf16_kernel(
    const float* __restrict__ src, unsigned short* __restrict__ dst, int K, int N) {
  int idx = blockIdx.x * blockDim.x + threadIdx.x;
  if (idx >= K * N) return;
  int k = idx / N, n = idx % N;
  dst[(size_t)n * K + k] = f32_to_bf16(src[idx]);
}

// ---------------------------------------------------------------------------
// Tiled bf16 WMMA GEMM:  C[M,N] = A[M,K] * Bt[N,K]^T + bias, fused epilogues.
// BM=BN=128, BK=32. 256 threads = 8 wave32, wave grid 4(M) x 2(N); each wave
// owns a 32x64 region = 2x4 16x16 accumulators -> 8 WMMA per wave per K-step.
// Double-buffered LDS (16KB A + 16KB B), one barrier per K-step. bf16 tiles
// (weights always; activations when already bf16) stream via async-to-LDS.
// A layouts: row-major [M,K] (f32 or bf16)  or  feature-map [B,C,128,128].
// ---------------------------------------------------------------------------
enum { LAY_ROW = 0, LAY_FMAP = 1 };
enum { OUT_F32 = 0, OUT_BF16 = 1, OUT_BF16_RELU = 2, OUT_BF16_MASK = 3 };

template <typename AT, int ALAYOUT, int OMODE>
__global__ __launch_bounds__(256) void gemm_wmma_kernel(
    const AT* __restrict__ A, const unsigned short* __restrict__ Bt,
    const float* __restrict__ bias, const unsigned char* __restrict__ mask,
    void* __restrict__ Cout, int K, int lda, int ldc) {
  __shared__ unsigned short sA[2][128][32];   // [m][k] bf16 tile (double buffered)
  __shared__ unsigned short sB[2][128][32];   // [n][k] bf16 tile (transposed W)

  const int tid  = threadIdx.x;
  const int m0   = blockIdx.x * 128;
  const int n0   = blockIdx.y * 128;
  const int lane = tid & 31;
  const int wv   = tid >> 5;
  const int wm   = wv & 3;        // 4 wave-rows of 32
  const int wn   = wv >> 2;       // 2 wave-cols of 64
  const int half = lane >> 4;
  const int lr   = lane & 15;

  v8f acc[2][4];
#pragma unroll
  for (int mt = 0; mt < 2; ++mt)
#pragma unroll
    for (int nt = 0; nt < 4; ++nt) acc[mt][nt] = {};

  size_t fm_base = 0;
  if constexpr (ALAYOUT == LAY_FMAP) {
    const int bimg = m0 >> 14;          // HW = 128*128 = 16384
    const int hw0  = m0 & 16383;
    fm_base = (size_t)bimg * (size_t)K * 16384u + (size_t)hw0;
  }

  // stage one 128x32 A tile + 128x32 B tile into LDS buffer `buf`
  auto stage = [&](int k0, int buf) {
    if constexpr (ALAYOUT == LAY_ROW) {
      const int r  = tid >> 1;            // 0..127
      const int c0 = (tid & 1) << 4;      // 0 or 16
      if constexpr (sizeof(AT) == 4) {    // f32 source: convert in regs
        const float* ap = (const float*)A + (size_t)(m0 + r) * (size_t)lda + (size_t)(k0 + c0);
        alignas(16) unsigned short tmp[16];
#pragma unroll
        for (int j = 0; j < 4; ++j) {
          const float4 f = ((const float4*)ap)[j];
          tmp[j * 4 + 0] = f32_to_bf16(f.x);
          tmp[j * 4 + 1] = f32_to_bf16(f.y);
          tmp[j * 4 + 2] = f32_to_bf16(f.z);
          tmp[j * 4 + 3] = f32_to_bf16(f.w);
        }
        uint4* dst = (uint4*)&sA[buf][r][c0];
        dst[0] = *(const uint4*)&tmp[0];
        dst[1] = *(const uint4*)&tmp[8];
        if (k0 + 32 < K) __builtin_prefetch(ap + 32, 0, 1);   // global_prefetch_b8
      } else {                            // bf16 source: async straight to LDS
        const unsigned short* ap =
            (const unsigned short*)A + (size_t)(m0 + r) * (size_t)lda + (size_t)(k0 + c0);
        async_copy_b128(&sA[buf][r][c0],     ap);
        async_copy_b128(&sA[buf][r][c0 + 8], ap + 8);
      }
    } else {
      // channel-major feature map: two adjacent channels per thread so the
      // transposed LDS store becomes a packed 2xbf16 b32 store.
      const int cp = (tid >> 4) << 1;     // channel pair 0,2,...,30
      const int r0 = (tid & 15) << 3;     // pixel 0,8,...,120
      const float* ap0 = (const float*)A + fm_base + (size_t)(k0 + cp) * 16384u + (size_t)r0;
      const float* ap1 = ap0 + 16384;
      const float4 f00 = ((const float4*)ap0)[0], f01 = ((const float4*)ap0)[1];
      const float4 f10 = ((const float4*)ap1)[0], f11 = ((const float4*)ap1)[1];
      const float a[8] = { f00.x, f00.y, f00.z, f00.w, f01.x, f01.y, f01.z, f01.w };
      const float b[8] = { f10.x, f10.y, f10.z, f10.w, f11.x, f11.y, f11.z, f11.w };
#pragma unroll
      for (int i = 0; i < 8; ++i)
        *(unsigned int*)&sA[buf][r0 + i][cp] = pack_bf16x2(a[i], b[i]);
    }
    {
      const int n  = tid >> 1;
      const int c0 = (tid & 1) << 4;
      const unsigned short* bp = Bt + (size_t)(n0 + n) * (size_t)K + (size_t)(k0 + c0);
      async_copy_b128(&sB[buf][n][c0],     bp);     // weights: always async
      async_copy_b128(&sB[buf][n][c0 + 8], bp + 8);
    }
  };

  stage(0, 0);
  async_wait_all();
  __syncthreads();

  const int nsteps = K >> 5;
  for (int s = 0; s < nsteps; ++s) {
    const int buf = s & 1;
    if (s + 1 < nsteps) stage((s + 1) << 5, buf ^ 1);   // async copies overlap WMMAs

    // ---- fragments per ISA 7.12.2 lane layout ----
    BfFrag af[2], bf[4];
    const int ca = half * 8;     // A: lanes 0-15 K 0-7/16-23, lanes 16-31 K 8-15/24-31
#pragma unroll
    for (int mt = 0; mt < 2; ++mt) {
      const int ar = wm * 32 + mt * 16 + lr;
#pragma unroll
      for (int j = 0; j < 8; ++j) {
        af[mt].u[j]     = sA[buf][ar][ca + j];
        af[mt].u[8 + j] = sA[buf][ar][16 + ca + j];
      }
    }
    const int cb = half * 8;     // B: lanes 0-15 K 0-7/16-23, lanes 16-31 K 8-15/24-31
#pragma unroll
    for (int nt = 0; nt < 4; ++nt) {
      const int bn = wn * 64 + nt * 16 + lr;
#pragma unroll
      for (int j = 0; j < 8; ++j) {
        bf[nt].u[j]     = sB[buf][bn][cb + j];
        bf[nt].u[8 + j] = sB[buf][bn][16 + cb + j];
      }
    }

#pragma unroll
    for (int mt = 0; mt < 2; ++mt)
#pragma unroll
      for (int nt = 0; nt < 4; ++nt)
        acc[mt][nt] = __builtin_amdgcn_wmma_f32_16x16x32_bf16(
            false, af[mt].v, false, bf[nt].v, (short)0, acc[mt][nt], false, false);

    async_wait_all();            // our async copies for buf^1 are in LDS
    __syncthreads();
  }

  // ---- epilogue: C/D layout VGPR g -> M = half*8 + g, N = lr ----
#pragma unroll
  for (int mt = 0; mt < 2; ++mt) {
    const int gmb = m0 + wm * 32 + mt * 16 + half * 8;
#pragma unroll
    for (int nt = 0; nt < 4; ++nt) {
      const int gn   = n0 + wn * 64 + nt * 16 + lr;
      const float bs = bias[gn];
#pragma unroll
      for (int g = 0; g < 8; ++g) {
        const int gm = gmb + g;
        float v = acc[mt][nt][g] + bs;
        if constexpr (OMODE == OUT_BF16_RELU) v = fmaxf(v, 0.f);
        if constexpr (OMODE == OUT_BF16_MASK) v *= (mask[gm] ? 1.f : 0.f);
        if constexpr (OMODE == OUT_F32)
          ((float*)Cout)[(size_t)gm * (size_t)ldc + gn] = v;
        else
          ((unsigned short*)Cout)[(size_t)gm * (size_t)ldc + gn] = f32_to_bf16(v);
      }
    }
  }
}

// ---------------------------------------------------------------------------
// Wave-per-query: offset head (tanh), weight head (softmax), bilinear gather
// of pf (channel-last bf16), masked weighted sum -> ctx into hcat[:,256:512].
// ---------------------------------------------------------------------------
__global__ __launch_bounds__(256) void sample_ctx_kernel(
    const float* __restrict__ queries, const float* __restrict__ ref_points,
    const unsigned char* __restrict__ mask,
    const float* __restrict__ Wo, const float* __restrict__ bo,
    const float* __restrict__ Ww, const float* __restrict__ bw,
    const unsigned short* __restrict__ pf,   // [B, 128, 128, 256] bf16
    unsigned short* __restrict__ hcat)       // [B*Q, 512]
{
  const int lane = threadIdx.x & 31;
  const int p    = blockIdx.x * (blockDim.x >> 5) + (threadIdx.x >> 5);  // 0..B*Q-1
  const int b    = p >> 14;                                              // Q = 16384

  // cooperative load of the 256-d query: 8 elems per lane (2 x b128)
  const float4* qp = (const float4*)(queries + (size_t)p * 256 + lane * 8);
  const float4 qa = qp[0], qb = qp[1];
  const float q[8] = { qa.x, qa.y, qa.z, qa.w, qb.x, qb.y, qb.z, qb.w };

  // 12 dot products (8 offsets + 4 weights), partial per lane
  float acc[12];
#pragma unroll
  for (int j = 0; j < 12; ++j) acc[j] = 0.f;
  const float* wo = Wo + (size_t)(lane * 8) * 8;
  const float* ww = Ww + (size_t)(lane * 8) * 4;
#pragma unroll
  for (int i = 0; i < 8; ++i) {
    const float x = q[i];
#pragma unroll
    for (int j = 0; j < 8; ++j) acc[j]     += x * wo[i * 8 + j];
#pragma unroll
    for (int j = 0; j < 4; ++j) acc[8 + j] += x * ww[i * 4 + j];
  }
  // wave32 xor-shuffle reduction (all lanes end with full sums)
#pragma unroll
  for (int j = 0; j < 12; ++j)
    for (int o = 16; o > 0; o >>= 1) acc[j] += __shfl_xor(acc[j], o, 32);

  float offv[8];
#pragma unroll
  for (int j = 0; j < 8; ++j) offv[j] = tanhf(acc[j] + bo[j]) * 0.05f;

  float wl[4], mx = -1e30f;
#pragma unroll
  for (int j = 0; j < 4; ++j) { wl[j] = acc[8 + j] + bw[j]; mx = fmaxf(mx, wl[j]); }
  float se = 0.f;
#pragma unroll
  for (int j = 0; j < 4; ++j) { wl[j] = expf(wl[j] - mx); se += wl[j]; }
  const float m   = mask[p] ? 1.f : 0.f;
  const float inv = m / fmaxf(se, 1e-6f);   // masked + renormalized
#pragma unroll
  for (int j = 0; j < 4; ++j) wl[j] *= inv;

  const float rx = ref_points[(size_t)p * 2 + 0];
  const float ry = ref_points[(size_t)p * 2 + 1];

  float ctx[8];
#pragma unroll
  for (int i = 0; i < 8; ++i) ctx[i] = 0.f;

#pragma unroll
  for (int s = 0; s < 4; ++s) {
    const float gx = fminf(fmaxf(rx + offv[s * 2 + 0], -1.2f), 1.2f);
    const float gy = fminf(fmaxf(ry + offv[s * 2 + 1], -1.2f), 1.2f);
    const float x  = (gx + 1.f) * 0.5f * 127.f;   // align_corners=True, W-1
    const float y  = (gy + 1.f) * 0.5f * 127.f;
    const float x0f = floorf(x), y0f = floorf(y);
    const int   x0  = (int)x0f,  y0  = (int)y0f;
    const float wx1 = x - x0f,   wy1 = y - y0f;
    const float cw[4] = { (1.f - wx1) * (1.f - wy1), wx1 * (1.f - wy1),
                          (1.f - wx1) * wy1,         wx1 * wy1 };
    const int xs[4] = { x0, x0 + 1, x0,     x0 + 1 };
    const int ys[4] = { y0, y0,     y0 + 1, y0 + 1 };
#pragma unroll
    for (int c4 = 0; c4 < 4; ++c4) {
      const int xi = xs[c4], yi = ys[c4];
      if (xi < 0 || xi >= 128 || yi < 0 || yi >= 128) continue;   // zeros padding
      const float wc = cw[c4] * wl[s];
      union { uint4 q; unsigned short u[8]; } d;                  // 8 bf16 = b128
      d.q = *(const uint4*)(pf + (((size_t)(b * 128 + yi) * 128 + xi) * 256) + lane * 8);
#pragma unroll
      for (int i = 0; i < 8; ++i) ctx[i] += wc * bf16_to_f32(d.u[i]);
    }
  }

  union { uint4 q; unsigned short u[8]; } o;
#pragma unroll
  for (int i = 0; i < 8; ++i) o.u[i] = f32_to_bf16(ctx[i]);
  *(uint4*)(hcat + (size_t)p * 512 + 256 + lane * 8) = o.q;
}

// ---------------------------------------------------------------------------
// Host launcher
// ---------------------------------------------------------------------------
extern "C" void kernel_launch(void* const* d_in, const int* in_sizes, int n_in,
                              void* d_out, int out_size, void* d_ws, size_t ws_size,
                              hipStream_t stream) {
  const float*         queries     = (const float*)d_in[0];
  const float*         ref_points  = (const float*)d_in[1];
  const float*         feature_map = (const float*)d_in[2];
  const unsigned char* valid_mask  = (const unsigned char*)d_in[3];
  const float* Wq  = (const float*)d_in[4];
  const float* bq  = (const float*)d_in[5];
  const float* Wf  = (const float*)d_in[6];
  const float* bfb = (const float*)d_in[7];
  const float* Wo  = (const float*)d_in[8];
  const float* bo  = (const float*)d_in[9];
  const float* Ww  = (const float*)d_in[10];
  const float* bw  = (const float*)d_in[11];
  const float* W1  = (const float*)d_in[12];
  const float* b1  = (const float*)d_in[13];
  const float* W2  = (const float*)d_in[14];
  const float* b2  = (const float*)d_in[15];
  (void)in_sizes; (void)n_in; (void)out_size; (void)ws_size;

  constexpr int Mrows = 4 * 16384;     // B*Q = 65536 query rows
  constexpr int Pix   = 4 * 128 * 128; // 65536 pixels

  char*  ws  = (char*)d_ws;
  size_t off = 0;
  auto alloc = [&](size_t bytes) -> void* {
    void* p = (void*)(ws + off);
    off += (bytes + 255) & ~(size_t)255;
    return p;
  };
  unsigned short* WqT  = (unsigned short*)alloc((size_t)256 * 256 * 2);
  unsigned short* WfT  = (unsigned short*)alloc((size_t)256 * 256 * 2);
  unsigned short* W1T  = (unsigned short*)alloc((size_t)256 * 512 * 2);
  unsigned short* W2T  = (unsigned short*)alloc((size_t)256 * 256 * 2);
  unsigned short* pfb  = (unsigned short*)alloc((size_t)Pix * 256 * 2);    // [B,H,W,HD]
  unsigned short* hcat = (unsigned short*)alloc((size_t)Mrows * 512 * 2);  // [pq | ctx]
  unsigned short* h1b  = (unsigned short*)alloc((size_t)Mrows * 256 * 2);

  // 1) weight prep: transpose + bf16
  transpose_bf16_kernel<<<(256 * 256 + 255) / 256, 256, 0, stream>>>(Wq, WqT, 256, 256);
  transpose_bf16_kernel<<<(256 * 256 + 255) / 256, 256, 0, stream>>>(Wf, WfT, 256, 256);
  transpose_bf16_kernel<<<(512 * 256 + 255) / 256, 256, 0, stream>>>(W1, W1T, 512, 256);
  transpose_bf16_kernel<<<(256 * 256 + 255) / 256, 256, 0, stream>>>(W2, W2T, 256, 256);

  const dim3 ggrid(Mrows / 128, 256 / 128);  // (512, 2)

  // 2) pf = 1x1 conv over feature_map -> channel-last bf16
  gemm_wmma_kernel<float, LAY_FMAP, OUT_BF16><<<ggrid, 256, 0, stream>>>(
      feature_map, WfT, bfb, nullptr, pfb, /*K=*/256, /*lda=*/0, /*ldc=*/256);

  // 3) pq = queries@Wq + bq, masked, -> hcat[:, 0:256]
  gemm_wmma_kernel<float, LAY_ROW, OUT_BF16_MASK><<<ggrid, 256, 0, stream>>>(
      queries, WqT, bq, valid_mask, hcat, /*K=*/256, /*lda=*/256, /*ldc=*/512);

  // 4) offsets + softmax weights + bilinear sampling -> hcat[:, 256:512]
  sample_ctx_kernel<<<Mrows / 8, 256, 0, stream>>>(
      queries, ref_points, valid_mask, Wo, bo, Ww, bw, pfb, hcat);

  // 5) h1 = relu(hcat@W1 + b1)
  gemm_wmma_kernel<unsigned short, LAY_ROW, OUT_BF16_RELU><<<ggrid, 256, 0, stream>>>(
      hcat, W1T, b1, nullptr, h1b, /*K=*/512, /*lda=*/512, /*ldc=*/256);

  // 6) out = h1@W2 + b2 (f32)
  gemm_wmma_kernel<unsigned short, LAY_ROW, OUT_F32><<<ggrid, 256, 0, stream>>>(
      h1b, W2T, b2, nullptr, d_out, /*K=*/256, /*lda=*/256, /*ldc=*/256);
}
